// MultiHeadAttn_71184787964283
// MI455X (gfx1250) — matrix operand order repaired and checked by reading, not compiled, not run on hardware
//
#include <hip/hip_runtime.h>
#include <hip/hip_bf16.h>
#include <stdint.h>

// ---------------------------------------------------------------------------
// MI455X (gfx1250) MHA + SparseNormer.
//  - all GEMMs on v_wmma_f32_16x16x32_bf16 (fp32 accumulate)
//  - attention K/V tiles DMA'd to LDS by the Tensor Data Mover
//    (tensor_load_to_lds, TENSORcnt), issued by wave 0, double-buffered
//  - W tiles in the projections via global_load_async_to_lds_b128 (ASYNCcnt)
//  - mask bit-packed (pre-inverted) by a wave32 __ballot kernel
// ---------------------------------------------------------------------------

typedef __bf16 bf16_t;
typedef __attribute__((ext_vector_type(16))) __bf16 v16bf;
typedef __attribute__((ext_vector_type(8)))  __bf16 v8bf;
typedef __attribute__((ext_vector_type(8)))  float  v8f;
typedef __attribute__((ext_vector_type(4)))  unsigned int v4u;
typedef __attribute__((ext_vector_type(8)))  int v8i;
typedef __attribute__((ext_vector_type(4)))  int v4i;

#define B_     4
#define NQ_    2048
#define SK_    2048
#define ISIZE_ 1024
#define HSIZE_ 1024
#define H_     16
#define D_     64
#define SKW_   (SK_ / 32)   // packed mask words per row

static __device__ inline v8f wmma_bf16(v16bf a, v16bf b, v8f c) {
  return __builtin_amdgcn_wmma_f32_16x16x32_bf16(false, a, false, b, (short)0, c,
                                                 false, false);
}

// A-operand (16x32 bf16, row-major, stride lda). ISA 7.12.2 layout.
static __device__ inline v16bf load_a_bf16(const bf16_t* base, int lda, int lane) {
  const int m = lane & 15;
  const int c = (lane < 16) ? 0 : 8;
  const bf16_t* p = base + (size_t)m * lda + c;
  v8bf lo = *(const v8bf*)(p);
  v8bf hi = *(const v8bf*)(p + 16);
  v16bf r;
#pragma unroll
  for (int i = 0; i < 8; ++i) { r[i] = lo[i]; r[i + 8] = hi[i]; }
  return r;
}

// A-operand from fp32 source with on-the-fly bf16 conversion.
static __device__ inline v16bf load_a_f32(const float* base, int lda, int lane) {
  const int m = lane & 15;
  const int c = (lane < 16) ? 0 : 8;
  const float* p = base + (size_t)m * lda + c;
  v8f lo = *(const v8f*)(p);
  v8f hi = *(const v8f*)(p + 16);
  v16bf r;
#pragma unroll
  for (int i = 0; i < 8; ++i) { r[i] = (bf16_t)lo[i]; r[i + 8] = (bf16_t)hi[i]; }
  return r;
}

// B-operand (32x16) from an [N][K] row-major source: col n contiguous in K.
static __device__ inline v16bf load_bT(const bf16_t* base, int ldb, int lane) {
  const int n  = lane & 15;
  const int kb = (lane < 16) ? 0 : 8;
  const bf16_t* p = base + (size_t)n * ldb + kb;
  v8bf lo = *(const v8bf*)(p);
  v8bf hi = *(const v8bf*)(p + 16);
  v16bf r;
#pragma unroll
  for (int i = 0; i < 8; ++i) { r[i] = lo[i]; r[i + 8] = hi[i]; }
  return r;
}

// Async global->LDS copy, 16 bytes, tracked by ASYNCcnt (CDNA5 §15.18 op 98).
static __device__ inline void async_ld128(uint32_t lds_byte_off, const void* gaddr) {
  asm volatile("global_load_async_to_lds_b128 %0, %1, off"
               :: "v"(lds_byte_off), "v"(gaddr) : "memory");
}
static __device__ inline void wait_async0() {
  asm volatile("s_wait_asynccnt 0x0" ::: "memory");
}

// Tensor Data Mover: 2-D strided tensor tile -> contiguous LDS (bf16 elems).
// D# layout per CDNA5 ISA ch.8: group0 {flags, lds_addr, global_addr, type=2},
// group1 {data_size, tensor_dim0/1, tile_dim0/1, dim0 stride}; groups 2/3 = 0
// (2-D tile: tile_dim2/3 = 0 means unused).
static __device__ inline void tdm_load_2d(uint32_t lds_addr, const void* gaddr,
                                          uint32_t dim0, uint32_t dim1,
                                          uint32_t stride0) {
  const unsigned long long ga = (unsigned long long)(uintptr_t)gaddr;
  v4u g0;
  g0[0] = 1u;                                                  // count=1 (valid)
  g0[1] = lds_addr;                                            // LDS byte addr
  g0[2] = (uint32_t)ga;                                        // addr[31:0]
  g0[3] = (uint32_t)((ga >> 32) & 0x01FFFFFFu) | (2u << 30);   // addr[56:32]|type=2
  v8i g1;
  g1[0] = (int)(1u << 16);                                     // data_size = 2B
  g1[1] = (int)((dim0 & 0xFFFFu) << 16);                       // tensor_dim0 lo16
  g1[2] = (int)(((dim0 >> 16) & 0xFFFFu) | ((dim1 & 0xFFFFu) << 16));
  g1[3] = (int)(((dim1 >> 16) & 0xFFFFu) | ((dim0 & 0xFFFFu) << 16)); // tile_dim0
  g1[4] = (int)(dim1 & 0xFFFFu);                               // tile_dim1 (dim2=0)
  g1[5] = (int)stride0;                                        // dim0 stride lo32
  g1[6] = 0;
  g1[7] = 0;
  v4i z4 = {0, 0, 0, 0};
#if __has_include(<hip/amd_detail/amd_gfx1250_TDM.h>)
  v8i z8 = {0, 0, 0, 0, 0, 0, 0, 0};
  __builtin_amdgcn_tensor_load_to_lds(g0, g1, z4, z4, z8, 0);  // clang-23 form
#else
  __builtin_amdgcn_tensor_load_to_lds(g0, g1, z4, z4, 0);      // ROCm 7.2 form
#endif
}

// ---------------------------------------------------------------------------
__global__ __launch_bounds__(256) void k_f32_to_bf16(const float* __restrict__ src,
                                                     bf16_t* __restrict__ dst, int n) {
  int i = blockIdx.x * blockDim.x + threadIdx.x;
  if (i < n) dst[i] = (bf16_t)src[i];
}

// Pack mask bytes -> 1 bit/key, PRE-INVERTED (1 = keep), via wave32 ballot.
__global__ __launch_bounds__(256) void k_packmask(const unsigned char* __restrict__ m,
                                                  uint32_t* __restrict__ pm, int nwords) {
  const int w    = (blockIdx.x * blockDim.x + threadIdx.x) >> 5;
  const int lane = threadIdx.x & 31;
  if (w < nwords) {
    unsigned char byte = m[(size_t)w * 32 + lane];
    unsigned long long bits = __ballot(byte != 0);
    if (lane == 0) pm[w] = ~(uint32_t)bits;
  }
}

// ---------------------------------------------------------------------------
// Projection GEMM: Out = oscale * (X fp32 [8192x1024]) @ W^T (bf16 [1024][1024])
// Block: 8 waves, 128(M) x 128(N); W tile (8 KB) async-staged, double-buffered.
// LAYOUT 0 -> bf16 [b][h][n][d]; LAYOUT 1 -> bf16 [b][h][d][n].
// ---------------------------------------------------------------------------
template <int LAYOUT>
static __device__ inline size_t out_off(int row, int col) {
  const int b = row >> 11, n = row & 2047;
  const int h = col >> 6,  d = col & 63;
  if (LAYOUT == 0)
    return (((size_t)(b * H_ + h) * NQ_) + n) * D_ + d;
  else
    return (((size_t)(b * H_ + h) * D_) + d) * (size_t)SK_ + n;
}

template <int LAYOUT>
__global__ __launch_bounds__(256) void k_proj(const float* __restrict__ X,
                                              const bf16_t* __restrict__ W,
                                              bf16_t* __restrict__ Out,
                                              float oscale) {
  extern __shared__ char dsm[];
  const uint32_t dynbase = __builtin_amdgcn_groupstaticsize();
  const int t    = threadIdx.x;
  const int lane = t & 31;
  const int wave = t >> 5;
  const int row0 = (blockIdx.x * 8 + wave) * 16;
  const int n0   = blockIdx.y * 128;
  const float*  xrow = X + (size_t)row0 * ISIZE_;
  const bf16_t* wblk = W + (size_t)n0 * ISIZE_;

  auto stageW = [&](int k0, int par) {   // W tile [128 n][32 k] = 8 KB
#pragma unroll
    for (int i = 0; i < 2; ++i) {
      const int c = t + i * 256;              // 512 x 16B chunks
      const int row = c >> 2, ch = c & 3;     // 4 chunks per 64B row
      async_ld128(dynbase + par * 8192 + row * 64 + ch * 16,
                  wblk + (size_t)row * ISIZE_ + k0 + ch * 8);
    }
  };

  v8f acc[8] = {};
  stageW(0, 0);
  wait_async0();
  __syncthreads();

  for (int k0 = 0; k0 < ISIZE_; k0 += 32) {
    const int par = (k0 >> 5) & 1;
    if (k0 + 32 < ISIZE_) stageW(k0 + 32, par ^ 1);

    __builtin_prefetch(xrow + k0 + 128, 0, 0);
    const bf16_t* wt = (const bf16_t*)(dsm + par * 8192);
    v16bf a = load_a_f32(xrow + k0, ISIZE_, lane);
#pragma unroll
    for (int j = 0; j < 8; ++j) {
      v16bf bm = load_bT(wt + (size_t)(16 * j) * 32, 32, lane);
      acc[j] = wmma_bf16(a, bm, acc[j]);
    }
    wait_async0();
    __syncthreads();
  }

  const int n  = lane & 15;
  const int mh = (lane >> 4) * 8;
#pragma unroll
  for (int j = 0; j < 8; ++j)
#pragma unroll
    for (int r = 0; r < 8; ++r)
      Out[out_off<LAYOUT>(row0 + mh + r, n0 + 16 * j + n)] =
          (bf16_t)(acc[j][r] * oscale);
}

// ---------------------------------------------------------------------------
// Attention. Block = 8 waves = 8 q-tiles of one (b,h), 128 q rows.
// Per 32-key chunk: K [32][64] + V^T [64][32] DMA'd by TDM (wave 0 issues),
// double-buffered in 2 x 8 KB LDS. Mask from pre-inverted bit-packed pm.
// ---------------------------------------------------------------------------
__global__ __launch_bounds__(256) void k_attn(const bf16_t* __restrict__ Qb,
                                              const bf16_t* __restrict__ Kb,
                                              const bf16_t* __restrict__ Vt,
                                              const uint32_t* __restrict__ pm,
                                              const float* __restrict__ sn_bias,
                                              bf16_t* __restrict__ Ob) {
  __shared__ bf16_t Pt[8][16 * 32];          // static: per-wave P relayout tile
  extern __shared__ char dsm[];              // dynamic: 2 x 8 KB stage buffers
  const uint32_t dynbase = __builtin_amdgcn_groupstaticsize();

  const int t    = threadIdx.x;
  const int lane = t & 31;
  const int wave = t >> 5;
  const int bh   = blockIdx.x >> 4;          // 16 blocks per (b,h)
  const int qblk = (blockIdx.x & 15) * 128;
  const int q0   = qblk + wave * 16;
  const int b    = bh >> 4;
  const int h    = bh & 15;

  const bf16_t* Qbase = Qb + ((size_t)bh * NQ_ + q0) * D_;
  const bf16_t* Kbase = Kb + (size_t)bh * SK_ * D_;
  const bf16_t* Vbase = Vt + (size_t)bh * D_ * SK_;

  // TDM staging: two 2-D descriptors per 32-key chunk, issued by wave 0 only
  // (TDM ignores EXEC; one DMA per workgroup). Completion via TENSORcnt.
  auto stage = [&](int sk0, int par) {
    if (wave == 0) {
      const uint32_t base = dynbase + par * 8192;
      tdm_load_2d(base,        Kbase + (size_t)sk0 * D_, D_, 32, D_);   // K tile
      tdm_load_2d(base + 4096, Vbase + sk0,              32, D_, SK_);  // V tile
    }
  };

  const float bias = sn_bias[0];             // Q pre-scaled by 1/8 in k_proj
  const v16bf a0 = load_a_bf16(Qbase + 0,  D_, lane);
  const v16bf a1 = load_a_bf16(Qbase + 32, D_, lane);
  const int n  = lane & 15;
  const int mh = (lane >> 4) * 8;

  // Pre-inverted packed-mask rows for this lane's 8 rows (stride SKW_ words).
  const uint32_t* pmr = pm + ((size_t)b * NQ_ + q0 + mh) * SKW_;

  v8f   accO[4] = {};
  float rs[8]   = {};

  stage(0, 0);
  __builtin_amdgcn_s_wait_tensorcnt(0);
  __syncthreads();

  for (int cg = 0; cg < SK_ / 128; ++cg) {   // groups of 4 x 32-key chunks
    uint4 mw[8];
#pragma unroll
    for (int r = 0; r < 8; ++r)
      mw[r] = *(const uint4*)(pmr + (size_t)r * SKW_ + cg * 4);

#pragma unroll
    for (int cc = 0; cc < 4; ++cc) {
      const int c   = cg * 4 + cc;
      const int par = c & 1;
      const int sk0 = c * 32;
      if (sk0 + 32 < SK_) stage(sk0 + 32, par ^ 1);

      const bf16_t* Ktile = (const bf16_t*)(dsm + par * 8192);
      const bf16_t* Vtile = (const bf16_t*)(dsm + par * 8192 + 4096);

#pragma unroll
      for (int half = 0; half < 2; ++half) {
        v16bf b0 = load_bT(Ktile + (size_t)(16 * half) * D_ + 0,  D_, lane);
        v16bf b1 = load_bT(Ktile + (size_t)(16 * half) * D_ + 32, D_, lane);
        v8f s = {};
        s = wmma_bf16(a0, b0, s);
        s = wmma_bf16(a1, b1, s);
#pragma unroll
        for (int r = 0; r < 8; ++r) {
          const uint32_t keep =
              ((&mw[r].x)[cc] >> (n + 16 * half)) & 1u;   // 1 = not masked
          float rl = fmaxf(s[r] + bias, 0.0f);
          float tv = rl * rl * (float)keep;
          rs[r] += tv;
          Pt[wave][(mh + r) * 32 + 16 * half + n] = (bf16_t)tv;
        }
      }
      asm volatile("s_wait_dscnt 0x0" ::: "memory");   // same-wave P RAW
      v16bf pa = load_a_bf16(&Pt[wave][0], 32, lane);
#pragma unroll
      for (int nt = 0; nt < 4; ++nt) {
        v16bf bv = load_bT(Vtile + (size_t)(nt * 16) * 32, 32, lane);
        accO[nt] = wmma_bf16(pa, bv, accO[nt]);
      }
      __builtin_amdgcn_s_wait_tensorcnt(0);   // next buffer DMA complete
      __syncthreads();                        // all waves done with current
    }
  }

  // Row-sum reduction across the 16 lanes of each half (wave32).
#pragma unroll
  for (int r = 0; r < 8; ++r) {
    float v = rs[r];
    v += __shfl_xor(v, 1, 32);
    v += __shfl_xor(v, 2, 32);
    v += __shfl_xor(v, 4, 32);
    v += __shfl_xor(v, 8, 32);
    rs[r] = 1.0f / (v + 1e-32f);
  }

  bf16_t* obase = Ob + ((size_t)b * NQ_ + q0) * HSIZE_ + h * D_;
#pragma unroll
  for (int nt = 0; nt < 4; ++nt)
#pragma unroll
    for (int r = 0; r < 8; ++r) {
      float o = accO[nt][r] * rs[r];
      obase[(size_t)(mh + r) * HSIZE_ + nt * 16 + n] = (bf16_t)o;
    }
}

// ---------------------------------------------------------------------------
// Output GEMM: out fp32 = Ob(bf16 [8192x1024]) @ Wo^T, same staging scheme.
// ---------------------------------------------------------------------------
__global__ __launch_bounds__(256) void k_ogemm(const bf16_t* __restrict__ A,
                                               const bf16_t* __restrict__ W,
                                               float* __restrict__ Out) {
  extern __shared__ char dsm[];
  const uint32_t dynbase = __builtin_amdgcn_groupstaticsize();
  const int t    = threadIdx.x;
  const int lane = t & 31;
  const int wave = t >> 5;
  const int row0 = (blockIdx.x * 8 + wave) * 16;
  const int n0   = blockIdx.y * 128;
  const bf16_t* arow = A + (size_t)row0 * HSIZE_;
  const bf16_t* wblk = W + (size_t)n0 * HSIZE_;

  auto stageW = [&](int k0, int par) {
#pragma unroll
    for (int i = 0; i < 2; ++i) {
      const int c = t + i * 256;
      const int row = c >> 2, ch = c & 3;
      async_ld128(dynbase + par * 8192 + row * 64 + ch * 16,
                  wblk + (size_t)row * HSIZE_ + k0 + ch * 8);
    }
  };

  v8f acc[8] = {};
  stageW(0, 0);
  wait_async0();
  __syncthreads();

  for (int k0 = 0; k0 < HSIZE_; k0 += 32) {
    const int par = (k0 >> 5) & 1;
    if (k0 + 32 < HSIZE_) stageW(k0 + 32, par ^ 1);

    __builtin_prefetch(arow + k0 + 128, 0, 0);
    const bf16_t* wt = (const bf16_t*)(dsm + par * 8192);
    v16bf a = load_a_bf16(arow + k0, HSIZE_, lane);
#pragma unroll
    for (int j = 0; j < 8; ++j) {
      v16bf bm = load_bT(wt + (size_t)(16 * j) * 32, 32, lane);
      acc[j] = wmma_bf16(a, bm, acc[j]);
    }
    wait_async0();
    __syncthreads();
  }

  const int n  = lane & 15;
  const int mh = (lane >> 4) * 8;
#pragma unroll
  for (int j = 0; j < 8; ++j)
#pragma unroll
    for (int r = 0; r < 8; ++r)
      Out[(size_t)(row0 + mh + r) * HSIZE_ + n0 + 16 * j + n] = acc[j][r];
}

// ---------------------------------------------------------------------------
extern "C" void kernel_launch(void* const* d_in, const int* in_sizes, int n_in,
                              void* d_out, int out_size, void* d_ws, size_t ws_size,
                              hipStream_t stream) {
  const float* iQ = (const float*)d_in[0];
  const float* iK = (const float*)d_in[1];
  const float* iV = (const float*)d_in[2];
  const unsigned char* mask = (const unsigned char*)d_in[3];  // jax bool = 1 byte
  const float* Wq = (const float*)d_in[4];
  const float* Wk = (const float*)d_in[5];
  const float* Wv = (const float*)d_in[6];
  const float* Wo = (const float*)d_in[7];
  const float* sn_bias = (const float*)d_in[8];
  float* out = (float*)d_out;

  // Workspace: 4 x 1M bf16 weights + 4 x 8M bf16 activations + 512K u32 mask.
  bf16_t* Wqb = (bf16_t*)d_ws;
  bf16_t* Wkb = Wqb + (1 << 20);
  bf16_t* Wvb = Wkb + (1 << 20);
  bf16_t* Wob = Wvb + (1 << 20);
  bf16_t* Qb  = Wob + (1 << 20);
  bf16_t* Kb  = Qb  + (8 << 20);
  bf16_t* Vtb = Kb  + (8 << 20);
  bf16_t* Ob  = Vtb + (8 << 20);
  uint32_t* pmask = (uint32_t*)(Ob + (8 << 20));

  const dim3 blk(256);
  const int NW = 1 << 20;
  k_f32_to_bf16<<<NW / 256, blk, 0, stream>>>(Wq, Wqb, NW);
  k_f32_to_bf16<<<NW / 256, blk, 0, stream>>>(Wk, Wkb, NW);
  k_f32_to_bf16<<<NW / 256, blk, 0, stream>>>(Wv, Wvb, NW);
  k_f32_to_bf16<<<NW / 256, blk, 0, stream>>>(Wo, Wob, NW);

  const int nwords = B_ * NQ_ * SKW_;                    // 524288
  k_packmask<<<(nwords * 32) / 256, blk, 0, stream>>>(mask, pmask, nwords);

  const dim3 pg(64, 8);   // 8192/128 x 1024/128
  k_proj<0><<<pg, blk, 16384, stream>>>(iQ, Wqb, Qb, 0.125f);  // fold 1/sqrt(D)
  k_proj<0><<<pg, blk, 16384, stream>>>(iK, Wkb, Kb, 1.0f);
  k_proj<1><<<pg, blk, 16384, stream>>>(iV, Wvb, Vtb, 1.0f);

  k_attn<<<1024, blk, 16384, stream>>>(Qb, Kb, Vtb, pmask, sn_bias, Ob);

  k_ogemm<<<pg, blk, 16384, stream>>>(Ob, Wob, out);
}
